// GCN_88734024335852
// MI455X (gfx1250) — matrix operand order repaired and checked by reading, not compiled, not run on hardware
//
#include <hip/hip_runtime.h>
#include <hip/hip_bf16.h>

typedef __attribute__((ext_vector_type(16))) _Float16 v16h;
typedef __attribute__((ext_vector_type(8)))  float    v8f;

#define HIDDEN 16
#define DFEAT  128

// ---------------- init: deg=1 (self loop), xe=0, agg1=0, out=0 ----------------
__global__ void k_init(float* __restrict__ deg, float* __restrict__ xe,
                       float* __restrict__ agg1, float* __restrict__ out, int n) {
  int i = blockIdx.x * blockDim.x + threadIdx.x;
  if (i < n) { deg[i] = 1.0f; xe[i] = 0.0f; out[i] = 0.0f; }
  if (i < n * HIDDEN) agg1[i] = 0.0f;
}

// ---------------- edge pre-pass: in-degree at col, edge_attr segment-sum at row
__global__ void k_edge_pre(const int* __restrict__ erow, const int* __restrict__ ecol,
                           const float* __restrict__ eattr,
                           float* __restrict__ deg, float* __restrict__ xe, int nE) {
  int e = blockIdx.x * blockDim.x + threadIdx.x;
  if (e >= nE) return;
  atomicAdd(&deg[ecol[e]], 1.0f);
  atomicAdd(&xe[erow[e]], eattr[e]);
}

__global__ void k_dinv(const float* __restrict__ deg, float* __restrict__ dinv, int n) {
  int i = blockIdx.x * blockDim.x + threadIdx.x;
  if (i < n) dinv[i] = rsqrtf(deg[i]);   // deg >= 1 always (self loop)
}

// ---------------- WMMA projection: hp1 = [x | xe] @ W1  (W1 is [129,16] row-major)
// One wave per 16-node tile. K=128 handled as 4 chunks of v_wmma_f32_16x16x32_f16,
// the 129th input column (xe) handled as a rank-1 VALU update.
__global__ __launch_bounds__(256) void k_gemm_wmma(
    const float* __restrict__ x, const float* __restrict__ xe,
    const float* __restrict__ W1, float* __restrict__ hp1, int n) {
  const int lane  = threadIdx.x & 31;
  const int half  = lane >> 4;          // 0: lanes 0-15, 1: lanes 16-31
  const int l     = lane & 15;
  const int wave  = blockIdx.x * (blockDim.x >> 5) + (threadIdx.x >> 5);
  const int nwave = gridDim.x * (blockDim.x >> 5);

  // Preload B fragments (32x16 K-chunks of W1). B k order used here (same as A):
  // lanes 0-15: N=l, K = kc*32 + 0..15 ; lanes 16-31: N=l, K = kc*32 + 16..31.
  v16h bfrag[4];
#pragma unroll
  for (int kc = 0; kc < 4; ++kc) {
    const int kb = kc * 32 + half * 16;
#pragma unroll
    for (int j = 0; j < 16; ++j)
      bfrag[kc][j] = (_Float16)W1[(kb + j) * HIDDEN + l];
  }
  const float wlast = W1[DFEAT * HIDDEN + l];     // W1[128][l] (xe column weight)

  const int nTiles = (n + 15) >> 4;
  for (int t = wave; t < nTiles; t += nwave) {
    int row = t * 16 + l;                       // A row index for this lane
    int rclamp = row < n ? row : (n - 1);       // keep EXEC full for WMMA
    const float* xr = x + (size_t)rclamp * DFEAT;

    v8f acc = {};
#pragma unroll
    for (int kc = 0; kc < 4; ++kc) {
      // A k order used here (16x32, same as B): lanes 0-15 hold K = 0..15 (+kc*32),
      // lanes 16-31 hold K = 16..31 (+kc*32).
      const int k1 = kc * 32 + half * 16;
      const int k2 = kc * 32 + half * 16 + 8;
      v16h a;
#pragma unroll
      for (int j = 0; j < 8; ++j) a[j]     = (_Float16)xr[k1 + j];
#pragma unroll
      for (int j = 0; j < 8; ++j) a[8 + j] = (_Float16)xr[k2 + j];
      acc = __builtin_amdgcn_wmma_f32_16x16x32_f16(
          /*neg_a=*/false, a, /*neg_b=*/false, bfrag[kc],
          /*c_mod=*/(short)0, acc, /*reuse_a=*/false, /*reuse_b=*/false);
    }

    // C/D layout: lane holds N=l, rows M = mbase + r in VGPR r.
    const int mbase = t * 16 + half * 8;
    if (t * 16 + 16 <= n) {
      // Wave-uniform full-tile fast path: branchless stores, vector xe loads.
      float4 xe0 = *(const float4*)(xe + mbase);      // rows mbase+0..3
      float4 xe1 = *(const float4*)(xe + mbase + 4);  // rows mbase+4..7
      float xev[8] = {xe0.x, xe0.y, xe0.z, xe0.w, xe1.x, xe1.y, xe1.z, xe1.w};
      float* hp = hp1 + (size_t)mbase * HIDDEN + l;
#pragma unroll
      for (int r = 0; r < 8; ++r)
        hp[r * HIDDEN] = acc[r] + xev[r] * wlast;
    } else {
#pragma unroll
      for (int r = 0; r < 8; ++r) {
        int m = mbase + r;
        if (m < n) hp1[(size_t)m * HIDDEN + l] = acc[r] + xe[m] * wlast;
      }
    }
  }
}

// ---------------- conv1 edge scatter: 16 lanes per edge (coalesced 64B gather+atomic)
__global__ void k_scatter1(const int* __restrict__ erow, const int* __restrict__ ecol,
                           const float* __restrict__ hp1, const float* __restrict__ dinv,
                           float* __restrict__ agg1, int nE) {
  long long tid = (long long)blockIdx.x * blockDim.x + threadIdx.x;
  int e = (int)(tid >> 4);
  int f = (int)(tid & 15);
  if (e >= nE) return;
  int r = erow[e], c = ecol[e];
  float nrm = dinv[r] * dinv[c];
  atomicAdd(&agg1[(size_t)c * HIDDEN + f], hp1[(size_t)r * HIDDEN + f] * nrm);
}

// ---------------- per node: self-loop, +b1, relu, project with W2 -> hp2
__global__ void k_node_mid(const float* __restrict__ agg1, const float* __restrict__ hp1,
                           const float* __restrict__ dinv, const float* __restrict__ b1,
                           const float* __restrict__ W2, float* __restrict__ hp2, int n) {
  int i = blockIdx.x * blockDim.x + threadIdx.x;
  if (i >= n) return;
  float di2 = dinv[i] * dinv[i];
  float acc = 0.0f;
#pragma unroll
  for (int f = 0; f < HIDDEN; ++f) {
    float h = agg1[(size_t)i * HIDDEN + f] + hp1[(size_t)i * HIDDEN + f] * di2 + b1[f];
    h = fmaxf(h, 0.0f);
    acc += h * W2[f];
  }
  hp2[i] = acc;
}

// ---------------- conv2 edge scatter: one atomic per edge
__global__ void k_scatter2(const int* __restrict__ erow, const int* __restrict__ ecol,
                           const float* __restrict__ hp2, const float* __restrict__ dinv,
                           float* __restrict__ out, int nE) {
  int e = blockIdx.x * blockDim.x + threadIdx.x;
  if (e >= nE) return;
  int r = erow[e], c = ecol[e];
  atomicAdd(&out[c], hp2[r] * dinv[r] * dinv[c]);
}

// ---------------- epilogue: self loop + b2
__global__ void k_final(const float* __restrict__ hp2, const float* __restrict__ dinv,
                        const float* __restrict__ b2, float* __restrict__ out, int n) {
  int i = blockIdx.x * blockDim.x + threadIdx.x;
  if (i >= n) return;
  out[i] += hp2[i] * dinv[i] * dinv[i] + b2[0];
}

extern "C" void kernel_launch(void* const* d_in, const int* in_sizes, int n_in,
                              void* d_out, int out_size, void* d_ws, size_t ws_size,
                              hipStream_t stream) {
  const float* x     = (const float*)d_in[0];   // [N,128]
  const float* eattr = (const float*)d_in[1];   // [E,1]
  const int*   erow  = (const int*)d_in[2];     // [E]
  const int*   ecol  = (const int*)d_in[3];     // [E]
  const float* W1    = (const float*)d_in[4];   // [129,16]
  const float* b1    = (const float*)d_in[5];   // [16]
  const float* W2    = (const float*)d_in[6];   // [16,1]
  const float* b2    = (const float*)d_in[7];   // [1]
  float* out = (float*)d_out;

  const int N = out_size;        // output is [N,1]
  const int E = in_sizes[2];

  // Workspace layout (floats): deg[N] dinv[N] xe[N] hp2[N] hp1[16N] agg1[16N] = 36N
  float* ws   = (float*)d_ws;
  float* deg  = ws;
  float* dinv = ws + (size_t)N;
  float* xe   = ws + 2 * (size_t)N;
  float* hp2  = ws + 3 * (size_t)N;
  float* hp1  = ws + 4 * (size_t)N;
  float* agg1 = ws + 20 * (size_t)N;

  const int T = 256;
  k_init<<<(N * HIDDEN + T - 1) / T, T, 0, stream>>>(deg, xe, agg1, out, N);
  k_edge_pre<<<(E + T - 1) / T, T, 0, stream>>>(erow, ecol, eattr, deg, xe, E);
  k_dinv<<<(N + T - 1) / T, T, 0, stream>>>(deg, dinv, N);

  int nTiles = (N + 15) / 16;
  int blocks = (nTiles + 7) / 8;             // 8 waves per 256-thread block, 1 tile/wave
  k_gemm_wmma<<<blocks, 256, 0, stream>>>(x, xe, W1, hp1, N);

  long long t1 = (long long)E * 16;
  k_scatter1<<<(unsigned)((t1 + T - 1) / T), T, 0, stream>>>(erow, ecol, hp1, dinv, agg1, E);
  k_node_mid<<<(N + T - 1) / T, T, 0, stream>>>(agg1, hp1, dinv, b1, W2, hp2, N);
  k_scatter2<<<(E + T - 1) / T, T, 0, stream>>>(erow, ecol, hp2, dinv, out, E);
  k_final<<<(N + T - 1) / T, T, 0, stream>>>(hp2, dinv, b2, out, N);
}
